// SinStyleMod_249108103688
// MI455X (gfx1250) — matrix operand order repaired and checked by reading, not compiled, not run on hardware
//
#include <hip/hip_runtime.h>
#include <hip/hip_bf16.h>

typedef __attribute__((ext_vector_type(16))) _Float16 v16h;
typedef __attribute__((ext_vector_type(8)))  float    v8f;
typedef __attribute__((ext_vector_type(4)))  float    v4f;

#define B_SZ   8
#define N_PTS  4096
#define IN_C   512
#define OUT_C  512
#define SDIM   512

// ---------------------------------------------------------------------------
// Kernel 1: s1[b][i] = 1 + mod_b[i] + dot(style[b,:], mod_w[i,:])
// ---------------------------------------------------------------------------
__global__ __launch_bounds__(256)
void style_fc_kernel(const float* __restrict__ style,
                     const float* __restrict__ mod_w,
                     const float* __restrict__ mod_b,
                     float* __restrict__ s1) {
    int idx = blockIdx.x * blockDim.x + threadIdx.x;   // b * IN_C
    int b = idx / IN_C;
    int i = idx - b * IN_C;
    const float* sv = style + (size_t)b * SDIM;
    const float* wv = mod_w + (size_t)i * SDIM;
    float acc = 0.f;
    #pragma unroll 4
    for (int k = 0; k < SDIM; k += 4) {
        v4f s = *(const v4f*)(sv + k);
        v4f w = *(const v4f*)(wv + k);
        acc += s[0]*w[0] + s[1]*w[1] + s[2]*w[2] + s[3]*w[3];
    }
    s1[idx] = acc + mod_b[i] + 1.0f;
}

// ---------------------------------------------------------------------------
// Kernel 2: demod[b][o] = rsqrt( sum_i (weight[i][o]*s1[b][i])^2 + eps )
// ---------------------------------------------------------------------------
__global__ __launch_bounds__(256)
void demod_kernel(const float* __restrict__ weight,
                  const float* __restrict__ s1,
                  float* __restrict__ demod) {
    int idx = blockIdx.x * blockDim.x + threadIdx.x;   // b * OUT_C
    int b = idx / OUT_C;
    int o = idx - b * OUT_C;
    const float* s = s1 + (size_t)b * IN_C;
    float acc = 0.f;
    #pragma unroll 4
    for (int i = 0; i < IN_C; ++i) {
        float w = weight[(size_t)i * OUT_C + o] * s[i];
        acc += w * w;
    }
    demod[idx] = rsqrtf(acc + 1e-8f);
}

// ---------------------------------------------------------------------------
// Kernel 3: pack Wt[b][o][i] = f16( weight[i][o] * s1[b][i] * demod[b][o] )
// LDS-tiled 32x32 transpose so both global read and write are coalesced.
// ---------------------------------------------------------------------------
__global__ __launch_bounds__(256)
void pack_w_kernel(const float* __restrict__ weight,
                   const float* __restrict__ s1,
                   const float* __restrict__ demod,
                   _Float16* __restrict__ wt) {
    __shared__ float tile[32][33];
    const int b  = blockIdx.z;
    const int i0 = blockIdx.x * 32;
    const int o0 = blockIdx.y * 32;
    const int c  = threadIdx.x & 31;
    const int r8 = threadIdx.x >> 5;    // 0..7

    #pragma unroll
    for (int p = 0; p < 4; ++p) {
        int r = r8 + p * 8;             // i offset within tile
        tile[r][c] = weight[(size_t)(i0 + r) * OUT_C + o0 + c]
                   * s1[(size_t)b * IN_C + i0 + r];
    }
    __syncthreads();
    #pragma unroll
    for (int p = 0; p < 4; ++p) {
        int r = r8 + p * 8;             // o offset within tile
        float d = demod[(size_t)b * OUT_C + o0 + r];
        wt[((size_t)b * OUT_C + o0 + r) * IN_C + i0 + c] =
            (_Float16)(tile[c][r] * d);
    }
}

// ---------------------------------------------------------------------------
// A-fragment load: 16x32 f16 layout. Lane l: m=l&15, h=l>>4;
// elems 0..15 <- K = k0+h*16+{0..15} (contiguous, matches B).
// Source rows are fp32; convert on the fly (v_cvt_pk_f16_f32, co-executes
// with XDL WMMA).
// ---------------------------------------------------------------------------
__device__ __forceinline__ v16h load_a_frag_f32(const float* __restrict__ pa) {
    v4f a0 = *(const v4f*)(pa);
    v4f a1 = *(const v4f*)(pa + 4);
    v4f a2 = *(const v4f*)(pa + 8);
    v4f a3 = *(const v4f*)(pa + 12);
    v16h af;
    af[0]  = (_Float16)a0[0]; af[1]  = (_Float16)a0[1];
    af[2]  = (_Float16)a0[2]; af[3]  = (_Float16)a0[3];
    af[4]  = (_Float16)a1[0]; af[5]  = (_Float16)a1[1];
    af[6]  = (_Float16)a1[2]; af[7]  = (_Float16)a1[3];
    af[8]  = (_Float16)a2[0]; af[9]  = (_Float16)a2[1];
    af[10] = (_Float16)a2[2]; af[11] = (_Float16)a2[3];
    af[12] = (_Float16)a3[0]; af[13] = (_Float16)a3[1];
    af[14] = (_Float16)a3[2]; af[15] = (_Float16)a3[3];
    return af;
}

// ---------------------------------------------------------------------------
// Kernel 4: batched GEMM  out[b] = x[b] (4096x512 f32) @ W[b] (512x512, f16 Wt)
// 256 threads = 8 waves. Block = 256(M) x 64(N); wave = 32(M) x 64(N).
// K-loop: 16 steps, 8 x v_wmma_f32_16x16x32_f16 per step sharing 4 B frags.
// ---------------------------------------------------------------------------
__global__ __launch_bounds__(256)
void modconv_wmma_gemm(const float* __restrict__ x,
                       const _Float16* __restrict__ wt,
                       float* __restrict__ out) {
    const int b    = blockIdx.z;
    const int m0   = blockIdx.x * 256;
    const int n0   = blockIdx.y * 64;
    const int wave = threadIdx.x >> 5;
    const int lane = threadIdx.x & 31;
    const int lm   = lane & 15;
    const int lh   = lane >> 4;

    // A: lane holds row m = lm of each 16-row strip (two strips per wave).
    const float* arow0 = x + (size_t)b * N_PTS * IN_C
                           + (size_t)(m0 + wave * 32 + lm) * IN_C;
    const float* arow1 = arow0 + (size_t)16 * IN_C;
    // B: lane holds column n (= row of Wt) for each of 4 N-subtiles;
    // per lane 16 contiguous K halves starting at k0 + lh*16.
    const _Float16* bbase = wt + (size_t)b * OUT_C * IN_C
                               + (size_t)(n0 + lm) * IN_C;

    v8f acc00 = {}, acc01 = {}, acc02 = {}, acc03 = {};
    v8f acc10 = {}, acc11 = {}, acc12 = {}, acc13 = {};

    for (int k0 = 0; k0 < IN_C; k0 += 32) {
        const int ka = k0 + lh * 16;
        const int kb = k0 + lh * 16;

        v16h af0 = load_a_frag_f32(arow0 + ka);
        v16h af1 = load_a_frag_f32(arow1 + ka);

        v16h bf0 = *(const v16h*)(bbase + 0 * 16 * IN_C + kb);
        v16h bf1 = *(const v16h*)(bbase + 1 * 16 * IN_C + kb);
        v16h bf2 = *(const v16h*)(bbase + 2 * 16 * IN_C + kb);
        v16h bf3 = *(const v16h*)(bbase + 3 * 16 * IN_C + kb);

        acc00 = __builtin_amdgcn_wmma_f32_16x16x32_f16(false, af0, false, bf0,
                                                       (short)0, acc00, false, false);
        acc01 = __builtin_amdgcn_wmma_f32_16x16x32_f16(false, af0, false, bf1,
                                                       (short)0, acc01, false, false);
        acc02 = __builtin_amdgcn_wmma_f32_16x16x32_f16(false, af0, false, bf2,
                                                       (short)0, acc02, false, false);
        acc03 = __builtin_amdgcn_wmma_f32_16x16x32_f16(false, af0, false, bf3,
                                                       (short)0, acc03, false, false);
        acc10 = __builtin_amdgcn_wmma_f32_16x16x32_f16(false, af1, false, bf0,
                                                       (short)0, acc10, false, false);
        acc11 = __builtin_amdgcn_wmma_f32_16x16x32_f16(false, af1, false, bf1,
                                                       (short)0, acc11, false, false);
        acc12 = __builtin_amdgcn_wmma_f32_16x16x32_f16(false, af1, false, bf2,
                                                       (short)0, acc12, false, false);
        acc13 = __builtin_amdgcn_wmma_f32_16x16x32_f16(false, af1, false, bf3,
                                                       (short)0, acc13, false, false);
    }

    // ---- Store C: VGPR r -> row lh*8 + r, col = lm (+16 per N-subtile).
    // out is write-once: use non-temporal stores so it doesn't evict x/Wt.
    const size_t obase = (size_t)b * N_PTS * OUT_C
                       + (size_t)(m0 + wave * 32 + lh * 8) * OUT_C
                       + n0 + lm;
    #pragma unroll
    for (int r = 0; r < 8; ++r) {
        float* p0 = out + obase + (size_t)r * OUT_C;
        __builtin_nontemporal_store(acc00[r], p0 + 0);
        __builtin_nontemporal_store(acc01[r], p0 + 16);
        __builtin_nontemporal_store(acc02[r], p0 + 32);
        __builtin_nontemporal_store(acc03[r], p0 + 48);
        float* p1 = p0 + (size_t)16 * OUT_C;
        __builtin_nontemporal_store(acc10[r], p1 + 0);
        __builtin_nontemporal_store(acc11[r], p1 + 16);
        __builtin_nontemporal_store(acc12[r], p1 + 32);
        __builtin_nontemporal_store(acc13[r], p1 + 48);
    }
}

// ---------------------------------------------------------------------------
// Host-side launch
// ---------------------------------------------------------------------------
extern "C" void kernel_launch(void* const* d_in, const int* in_sizes, int n_in,
                              void* d_out, int out_size, void* d_ws, size_t ws_size,
                              hipStream_t stream) {
    const float* x      = (const float*)d_in[0];   // (8, 4096, 512)
    const float* style  = (const float*)d_in[1];   // (8, 512)
    const float* weight = (const float*)d_in[2];   // (1, 512, 512)
    const float* mod_w  = (const float*)d_in[3];   // (512, 512)
    const float* mod_b  = (const float*)d_in[4];   // (512,)
    float* out = (float*)d_out;                    // (8, 4096, 512) f32

    char* ws = (char*)d_ws;
    float*     s1    = (float*)(ws);                // 8*512*4  = 16 KB
    float*     demod = (float*)(ws + (16 << 10));   // 16 KB
    _Float16*  wtf16 = (_Float16*)(ws + (32 << 10));// 8*512*512*2 = 4 MB

    style_fc_kernel<<<(B_SZ * IN_C) / 256, 256, 0, stream>>>(style, mod_w, mod_b, s1);
    demod_kernel<<<(B_SZ * OUT_C) / 256, 256, 0, stream>>>(weight, s1, demod);

    dim3 gp(IN_C / 32, OUT_C / 32, B_SZ);           // (16,16,8)
    pack_w_kernel<<<gp, 256, 0, stream>>>(weight, s1, demod, wtf16);

    dim3 gg(N_PTS / 256, OUT_C / 64, B_SZ);         // (16,8,8)
    modconv_wmma_gemm<<<gg, 256, 0, stream>>>(x, wtf16, out);
}
